// CharLSTM_66511863546041
// MI455X (gfx1250) — matrix operand order repaired and checked by reading, not compiled, not run on hardware
//
#include <hip/hip_runtime.h>
#include <hip/hip_bf16.h>

// ---------------------------------------------------------------------------
// CharLSTM on MI455X (gfx1250): persistent-kernel scan, bf16 WMMA GEMVs,
// L2-resident packed weights, split-K across waves, 1 grid barrier per step.
// 128 blocks -> 128 WGPs pulling from L2 concurrently.
// ---------------------------------------------------------------------------

#define N_CHARS 256
#define EMB     512
#define HID     2048
#define CAT     (EMB + HID)       // 2560
#define SEQ     8192
#define NBLK    128               // persistent blocks (co-resident, 1/WGP)
#define TPB     256               // 8 waves of 32
#define KT_G    (CAT / 32)        // 80 k-tiles total per gate GEMV
#define KT_H    (KT_G / 2)        // 40 k-tiles per wave (split-K by 2)
#define HPB     (HID / NBLK)      // 16 hidden units per block

typedef __attribute__((ext_vector_type(16))) __bf16 v16bf;
typedef __attribute__((ext_vector_type(8)))  __bf16 bf8;
typedef __attribute__((ext_vector_type(8)))  float  v8f;

// ---- workspace layout (bytes), all offsets 256-aligned -----------------
#define OFF_PW    ((size_t)0)                         // packed gate W, bf16: 8192*2560*2 = 41,943,040
#define OFF_WOUT  ((size_t)41943040)                  // Wout bf16 row-major: 256*2048*2 = 1,048,576
#define OFF_XEMB  ((size_t)42991616)                  // gathered emb bf16:  8192*512*2 = 8,388,608
#define OFF_HBUF  ((size_t)51380224)                  // h ping-pong bf16:   2*2048*2   = 8,192
#define OFF_BAR   ((size_t)51388416)                  // 2 x u32 barrier state

__global__ void zero_barrier(unsigned* bar) {
    bar[0] = 0u; bar[1] = 0u;
}

// Repack concat(Wf,Wi,Wo,Wc) [4][2048][2560] fp32 into WMMA-B fragment order:
// flat id = (((n_tile*KT_G + k_tile)*32 + lane)*16 + e)
// lane = (n%16) + 16*(k%32 >= 16);  e = k%16  (B 32x16: lanes 0-15 K0..15, 16-31 K16..31)
__global__ void pack_gates(const float* __restrict__ Wf, const float* __restrict__ Wi,
                           const float* __restrict__ Wo, const float* __restrict__ Wc,
                           __bf16* __restrict__ pw) {
    size_t id = (size_t)blockIdx.x * TPB + threadIdx.x;   // < 8192*2560
    int e    = (int)(id & 15);
    int lane = (int)((id >> 4) & 31);
    int kt   = (int)((id >> 9) % KT_G);
    int nt   = (int)(id / (512ull * KT_G));               // 0..511
    int n    = nt * 16 + (lane & 15);                     // global gate-output index 0..8191
    int k    = kt * 32 + ((lane >> 4) << 4) + e;          // 0..2559
    int g    = n >> 11;                                   // gate select
    int j    = n & 2047;                                  // row within gate
    const float* W = (g == 0) ? Wf : (g == 1) ? Wi : (g == 2) ? Wo : Wc;
    pw[id] = (__bf16)W[(size_t)j * CAT + k];
}

__global__ void pack_wout(const float* __restrict__ Wout, __bf16* __restrict__ wb) {
    size_t id = (size_t)blockIdx.x * TPB + threadIdx.x;   // < 256*2048
    wb[id] = (__bf16)Wout[id];
}

__global__ void gather_emb(const int* __restrict__ idx, const float* __restrict__ emb,
                           __bf16* __restrict__ xe) {
    size_t id = (size_t)blockIdx.x * TPB + threadIdx.x;   // < 8192*512
    int t = (int)(id / EMB), e = (int)(id % EMB);
    xe[id] = (__bf16)emb[(size_t)idx[t] * EMB + e];
}

__device__ __forceinline__ void grid_barrier(unsigned* cnt, unsigned* gen) {
    __threadfence();
    __syncthreads();
    if (threadIdx.x == 0) {
        unsigned g = ((volatile unsigned*)gen)[0];
        if (atomicAdd(cnt, 1u) == NBLK - 1) {
            ((volatile unsigned*)cnt)[0] = 0u;
            __threadfence();
            atomicAdd(gen, 1u);
        } else {
            while (((volatile unsigned*)gen)[0] == g) __builtin_amdgcn_s_sleep(1);
        }
    }
    __syncthreads();
}

__device__ __forceinline__ float sigm(float x) { return 1.0f / (1.0f + __expf(-x)); }

// vectorized bf16 dot over HID elements, one wave; returns full sum on lane 0
__device__ __forceinline__ float wave_dot_hid(const __bf16* __restrict__ a,
                                              const __bf16* __restrict__ b, int lane) {
    float sum = 0.0f;
    #pragma unroll
    for (int q = 0; q < HID / (32 * 8); ++q) {       // 8 chunks of 8 bf16 per lane
        int base = (q * 32 + lane) * 8;
        bf8 av = *(const bf8*)(a + base);
        bf8 bv = *(const bf8*)(b + base);
        #pragma unroll
        for (int e = 0; e < 8; ++e) sum += (float)av[e] * (float)bv[e];
    }
    #pragma unroll
    for (int off = 16; off; off >>= 1) sum += __shfl_xor(sum, off, 32);
    return sum;
}

__global__ __launch_bounds__(TPB, 1)
void lstm_persist(const __bf16* __restrict__ pw, const __bf16* __restrict__ woutb,
                  const __bf16* __restrict__ xemb, __bf16* __restrict__ hbuf,
                  const float* __restrict__ bf_, const float* __restrict__ bi_,
                  const float* __restrict__ bo_, const float* __restrict__ bc_,
                  const float* __restrict__ bout, const float* __restrict__ h0,
                  const float* __restrict__ c0, float* __restrict__ out,
                  unsigned* __restrict__ bar) {
    __shared__ __bf16 zsm[CAT];        // z = [x_t ; h_t]
    __shared__ float  psm[8][16];      // per-wave split-K partial sums (16 outputs each)

    const int tid  = threadIdx.x, b = blockIdx.x;
    const int w    = tid >> 5, lane = tid & 31;
    const int g    = w & 3;                                  // gate 0..3
    const int ks   = w >> 2;                                 // K slice 0/1 (40 k-tiles each)
    const int ntile = g * 128 + b;                           // this wave's 16-output tile
    const __bf16* pbase = pw + ((size_t)ntile * KT_G + (size_t)ks * KT_H) * 512;
    const float* bias_g = (g == 0) ? bf_ : (g == 1) ? bi_ : (g == 2) ? bo_ : bc_;

    // --- init h0 (bf16 ping buffer 0) and per-lane cell state (wave 0 lanes<16)
    float creg = 0.0f;
    if (w == 0 && lane < HPB) {
        int j = b * HPB + lane;
        creg = c0[j];
        hbuf[j] = (__bf16)h0[j];
    }
    grid_barrier(&bar[0], &bar[1]);

    const int off0 = (lane >= 16) ? 16 : 0;  // A-frag element offset (in bf16), contiguous k = 16h + s
    const int kbase = ks * KT_H * 32;        // z offset of this wave's K slice

    for (int s = 0; s < SEQ; ++s) {
        const int rp = s & 1, wp = rp ^ 1;

        // stage z = [xemb[s]; h_s] into LDS
        {
            const __bf16* xr = xemb + (size_t)s * EMB;
            for (int i = tid; i < EMB; i += TPB) zsm[i] = xr[i];
            const __bf16* hr = hbuf + (size_t)rp * HID;
            for (int i = tid; i < HID; i += TPB) zsm[EMB + i] = hr[i];
        }
        __syncthreads();

        // --- gate GEMV slice: 16 outputs, 40 k-tiles of wmma_f32_16x16x32_bf16
        v8f acc = {};
        for (int kt = 0; kt < KT_H; ++kt) {
            // A: broadcast z[kbase+kt*32 .. +31] into all 16 rows (two ds_load_b128)
            bf8 alo = *(const bf8*)&zsm[kbase + kt * 32 + off0];
            bf8 ahi = *(const bf8*)&zsm[kbase + kt * 32 + off0 + 8];
            v16bf a;
            #pragma unroll
            for (int e = 0; e < 8; ++e) { a[e] = alo[e]; a[8 + e] = ahi[e]; }
            // B: packed fragment, contiguous 32B per lane (two global_load_b128)
            const __bf16* fp = pbase + ((size_t)kt * 32 + lane) * 16;
            bf8 blo = *(const bf8*)fp;
            bf8 bhi = *(const bf8*)(fp + 8);
            v16bf bb;
            #pragma unroll
            for (int e = 0; e < 8; ++e) { bb[e] = blo[e]; bb[8 + e] = bhi[e]; }
            acc = __builtin_amdgcn_wmma_f32_16x16x32_bf16(
                      false, a, false, bb, (short)0, acc, false, false);
        }
        if (lane < 16) psm[w][lane] = acc[0];   // row 0 of D = the real partial dots

        // --- Wout @ h_s -> pred[s-1] (waves 4,5; h_s stable this whole step)
        if ((w == 4 || w == 5) && s >= 1) {
            int r = b * 2 + (w - 4);
            float sum = wave_dot_hid(woutb + (size_t)r * HID, hbuf + (size_t)rp * HID, lane);
            if (lane == 0) out[(size_t)(s - 1) * N_CHARS + r] = sum + bout[r];
        }
        __syncthreads();

        // --- pointwise LSTM update for this block's 16 hidden units (wave 0, lanes<16)
        if (w == 0 && lane < HPB) {
            int j = b * HPB + lane;
            float f  = sigm(psm[0][lane] + psm[4][lane] + bf_[j]);
            float ii = sigm(psm[1][lane] + psm[5][lane] + bi_[j]);
            float o  = sigm(psm[2][lane] + psm[6][lane] + bo_[j]);
            float ct = tanhf(psm[3][lane] + psm[7][lane] + bc_[j]);
            creg = f * creg + ii * ct;
            float h = o * tanhf(creg);
            hbuf[(size_t)wp * HID + j] = (__bf16)h;
            if (s == SEQ - 1) {   // final h, c of the tuple output
                out[(size_t)SEQ * N_CHARS + j] = h;
                out[(size_t)SEQ * N_CHARS + HID + j] = creg;
            }
        }
        grid_barrier(&bar[0], &bar[1]);
    }
    (void)bias_g;

    // --- last prediction: Wout @ h_SEQ (lives in buffer SEQ%2 == 0)
    if (w == 4 || w == 5) {
        int r = b * 2 + (w - 4);
        float sum = wave_dot_hid(woutb + (size_t)r * HID,
                                 hbuf + (size_t)(SEQ & 1) * HID, lane);
        if (lane == 0) out[(size_t)(SEQ - 1) * N_CHARS + r] = sum + bout[r];
    }
}

extern "C" void kernel_launch(void* const* d_in, const int* in_sizes, int n_in,
                              void* d_out, int out_size, void* d_ws, size_t ws_size,
                              hipStream_t stream) {
    const int*   idx  = (const int*)  d_in[0];
    const float* h0   = (const float*)d_in[1];
    const float* c0   = (const float*)d_in[2];
    const float* emb  = (const float*)d_in[3];
    const float* Wf   = (const float*)d_in[4];
    const float* bf_  = (const float*)d_in[5];
    const float* Wi   = (const float*)d_in[6];
    const float* bi_  = (const float*)d_in[7];
    const float* Wo   = (const float*)d_in[8];
    const float* bo_  = (const float*)d_in[9];
    const float* Wc   = (const float*)d_in[10];
    const float* bc_  = (const float*)d_in[11];
    const float* Wout = (const float*)d_in[12];
    const float* bout = (const float*)d_in[13];

    char* ws = (char*)d_ws;
    __bf16*   pw    = (__bf16*)(ws + OFF_PW);
    __bf16*   woutb = (__bf16*)(ws + OFF_WOUT);
    __bf16*   xemb  = (__bf16*)(ws + OFF_XEMB);
    __bf16*   hbuf  = (__bf16*)(ws + OFF_HBUF);
    unsigned* bar   = (unsigned*)(ws + OFF_BAR);

    zero_barrier<<<1, 1, 0, stream>>>(bar);
    pack_gates<<<(8192u * CAT) / TPB, TPB, 0, stream>>>(Wf, Wi, Wo, Wc, pw);
    pack_wout<<<(N_CHARS * HID) / TPB, TPB, 0, stream>>>(Wout, woutb);
    gather_emb<<<(SEQ * EMB) / TPB, TPB, 0, stream>>>(idx, emb, xemb);
    lstm_persist<<<NBLK, TPB, 0, stream>>>(pw, woutb, xemb, hbuf,
                                           bf_, bi_, bo_, bc_, bout,
                                           h0, c0, (float*)d_out, bar);
}
